// HEPOSCrossAttention_75445395522271
// MI455X (gfx1250) — matrix operand order repaired and checked by reading, not compiled, not run on hardware
//
#include <hip/hip_runtime.h>
#include <hip/hip_bf16.h>
#include <math.h>

// ---------------------------------------------------------------------------
// HEPOS cross-attention for MI455X (gfx1250, wave32, WMMA bf16).
// Phase 0: one-shot fp32 -> bf16 conversion of activations + weights.
// Phase 1: Q/K/V projections as pure-bf16 WMMA GEMMs (V stored transposed &
//          stride-compacted so attention P*V B-fragments are contiguous).
// Phase 2: flash attention, one wave per (b, h, 16-query tile).
// Phase 3: output projection back to fp32.
// ---------------------------------------------------------------------------

#define D_MODEL   1024
#define NUM_HEADS 16
#define HEAD_DIM  64
#define B_SZ      4
#define T_LEN     1024
#define S_LEN     4096
// sum over 16 heads of S/stride, strides = [1,2,4,8]*4 -> 4*(4096+2048+1024+512)
#define SUM_SH    30720

typedef __attribute__((ext_vector_type(16))) __bf16 v16bf;
typedef __attribute__((ext_vector_type(8)))  float  v8f;

union Frag {
    v16bf v;
    unsigned int u[8];
    int4 q[2];
};

__device__ __forceinline__ unsigned short bf16s(float x) {
    __bf16 h = (__bf16)x;                       // RNE via hardware cvt
    return __builtin_bit_cast(unsigned short, h);
}
__device__ __forceinline__ unsigned int bf16pk(float lo, float hi) {
    return (unsigned int)bf16s(lo) | ((unsigned int)bf16s(hi) << 16);
}
__device__ __forceinline__ v8f wmma_bf16(const Frag& a, const Frag& b, v8f c) {
    return __builtin_amdgcn_wmma_f32_16x16x32_bf16(
        /*neg_a=*/false, a.v, /*neg_b=*/false, b.v,
        /*c_mod=*/(short)0, c, /*reuse_a=*/false, /*reuse_b=*/false);
}

// Vt (stride-compacted, transposed V) offset of head h, batch b (in elements).
__device__ __forceinline__ size_t vt_off(int b, int h) {
    int pref = 0;
    for (int i = 0; i < h; ++i) pref += S_LEN >> (i & 3);
    return ((size_t)b * SUM_SH + (size_t)pref) * (size_t)HEAD_DIM;
}

// ---------------------------------------------------------------------------
// One-shot fp32 -> bf16 conversion, 8 elements / thread, b128 in / b128 out.
// ---------------------------------------------------------------------------
__global__ __launch_bounds__(256) void cvt_bf16_kernel(
    const float* __restrict__ src, unsigned short* __restrict__ dst, int n8)
{
    const int i = blockIdx.x * blockDim.x + threadIdx.x;
    if (i >= n8) return;
    const float4* s = (const float4*)src + (size_t)i * 2;
    float4 a = s[0], b = s[1];
    union { unsigned int u[4]; int4 q; } o;
    o.u[0] = bf16pk(a.x, a.y); o.u[1] = bf16pk(a.z, a.w);
    o.u[2] = bf16pk(b.x, b.y); o.u[3] = bf16pk(b.z, b.w);
    *((int4*)dst + i) = o.q;
}

// ---------------------------------------------------------------------------
// Projection GEMM (all-bf16 operands): Y = X * W^T + bias.
// One wave -> 16x64 output tile (n-group == head). mode: 0=Q, 1=K (store
// [B][H][rows][64] bf16), 2=V (store transposed + stride-compacted
// [B][H][64][rows>>stride_log] bf16).
// ---------------------------------------------------------------------------
__global__ __launch_bounds__(256) void gemm_qkv_kernel(
    const unsigned short* __restrict__ X, const unsigned short* __restrict__ W,
    const float* __restrict__ bias, unsigned short* __restrict__ out,
    int rowsLog2, int mode)
{
    const int lane = threadIdx.x & 31;
    const int half = lane >> 4;
    const int lm   = lane & 15;
    const int wave = blockIdx.x * (blockDim.x >> 5) + (threadIdx.x >> 5);
    const int rowsPerBatch = 1 << rowsLog2;
    const int mTiles = (B_SZ << rowsLog2) >> 4;
    if (wave >= mTiles * NUM_HEADS) return;
    const int h     = wave & 15;
    const int mBase = (wave >> 4) << 4;
    const int oBase = h << 6;
    const int aoff  = half * 16;

    v8f acc[4] = {};

    const unsigned short* arow = X + (size_t)(mBase + lm) * D_MODEL;
    const unsigned short* wrow0 = W + (size_t)(oBase + lm)      * D_MODEL + half*16;
    const unsigned short* wrow1 = W + (size_t)(oBase + 16 + lm) * D_MODEL + half*16;
    const unsigned short* wrow2 = W + (size_t)(oBase + 32 + lm) * D_MODEL + half*16;
    const unsigned short* wrow3 = W + (size_t)(oBase + 48 + lm) * D_MODEL + half*16;

    for (int k = 0; k < D_MODEL; k += 32) {
        __builtin_prefetch(arow + k + 32 + aoff, 0, 3);
        Frag a;
        a.q[0] = *(const int4*)(arow + k + aoff);
        a.q[1] = *(const int4*)(arow + k + 8 + aoff);
        Frag b0, b1, b2, b3;
        b0.q[0] = *(const int4*)(wrow0 + k); b0.q[1] = *(const int4*)(wrow0 + k + 8);
        b1.q[0] = *(const int4*)(wrow1 + k); b1.q[1] = *(const int4*)(wrow1 + k + 8);
        b2.q[0] = *(const int4*)(wrow2 + k); b2.q[1] = *(const int4*)(wrow2 + k + 8);
        b3.q[0] = *(const int4*)(wrow3 + k); b3.q[1] = *(const int4*)(wrow3 + k + 8);
        acc[0] = wmma_bf16(a, b0, acc[0]);
        acc[1] = wmma_bf16(a, b1, acc[1]);
        acc[2] = wmma_bf16(a, b2, acc[2]);
        acc[3] = wmma_bf16(a, b3, acc[3]);
    }

    #pragma unroll
    for (int j = 0; j < 4; ++j) {
        const float bb = bias[oBase + j*16 + lm];
        #pragma unroll
        for (int v = 0; v < 8; ++v) {
            const float val = acc[j][v] + bb;
            const int r = mBase + v + 8*half;
            const int b = r >> rowsLog2;
            const int s = r & (rowsPerBatch - 1);
            if (mode == 2) {
                const int sl = h & 3;                 // log2(stride)
                if ((s & ((1 << sl) - 1)) == 0) {
                    const int Sh = rowsPerBatch >> sl;
                    out[vt_off(b, h) + (size_t)(j*16 + lm) * Sh + (s >> sl)] =
                        bf16s(val);
                }
            } else {
                out[(((size_t)(b*NUM_HEADS + h)) * rowsPerBatch + s) * HEAD_DIM
                    + j*16 + lm] = bf16s(val);
            }
        }
    }
}

// ---------------------------------------------------------------------------
// Flash attention: one wave per (b, h, 16-query tile). K laid out
// [B][H][S][64] bf16 (strided rows read directly); V laid out transposed +
// compacted [B][H][64][Sh] bf16 so P*V B-fragments are contiguous loads.
// P tile converted C-layout -> A-layout via LDS.
// Writes bf16 attn output [B][T][D_MODEL] (heads concatenated).
// ---------------------------------------------------------------------------
__global__ __launch_bounds__(256) void attn_kernel(
    const unsigned short* __restrict__ Q, const unsigned short* __restrict__ K,
    const unsigned short* __restrict__ Vt, unsigned short* __restrict__ Out)
{
    __shared__ __attribute__((aligned(16))) unsigned short pbuf[8 * 16 * 32];
    const int lane = threadIdx.x & 31;
    const int half = lane >> 4;
    const int lm   = lane & 15;
    const int widb = threadIdx.x >> 5;
    const int wave = blockIdx.x * 8 + widb;
    const int qt   = wave & 63;            // T/16 = 64 query tiles
    const int h    = (wave >> 6) & 15;
    const int b    = wave >> 10;
    const int sl   = h & 3;
    const int Sh   = S_LEN >> sl;
    const int stride = 1 << sl;
    const int aoff = half * 16;

    unsigned short* pw = pbuf + widb * (16 * 32);

    // Q fragments (K-dim 64 -> two 16x32 A fragments), resident for the loop
    const unsigned short* qrow =
        Q + ((size_t)(b*NUM_HEADS + h) * T_LEN + qt*16 + lm) * HEAD_DIM;
    Frag qa[2];
    #pragma unroll
    for (int kk = 0; kk < 2; ++kk) {
        qa[kk].q[0] = *(const int4*)(qrow + kk*32 + aoff);
        qa[kk].q[1] = *(const int4*)(qrow + kk*32 + 8 + aoff);
    }

    const unsigned short* kbase = K + (size_t)(b*NUM_HEADS + h) * S_LEN * HEAD_DIM;
    const unsigned short* vbase = Vt + vt_off(b, h);

    v8f oacc[4] = {};
    float rmax[8], rsum[8];
    #pragma unroll
    for (int v = 0; v < 8; ++v) { rmax[v] = -3.0e38f; rsum[v] = 0.f; }
    const float scale = 0.125f;            // 1/sqrt(64)

    for (int sb = 0; sb < Sh; sb += 32) {
        // ---- scores: two 16x16 tiles (keys sb..sb+15, sb+16..sb+31) ----
        v8f c0 = {}, c1 = {};
        #pragma unroll
        for (int kk = 0; kk < 2; ++kk) {
            Frag b0, b1;
            const unsigned short* kr0 =
                kbase + (size_t)(sb + lm) * stride * HEAD_DIM + kk*32 + half*16;
            const unsigned short* kr1 =
                kbase + (size_t)(sb + 16 + lm) * stride * HEAD_DIM + kk*32 + half*16;
            b0.q[0] = *(const int4*)(kr0); b0.q[1] = *(const int4*)(kr0 + 8);
            b1.q[0] = *(const int4*)(kr1); b1.q[1] = *(const int4*)(kr1 + 8);
            c0 = wmma_bf16(qa[kk], b0, c0);
            c1 = wmma_bf16(qa[kk], b1, c1);
        }
        // ---- online softmax (rows live in one VGPR across 16 lanes) ----
        #pragma unroll
        for (int v = 0; v < 8; ++v) {
            float s0 = c0[v] * scale;
            float s1 = c1[v] * scale;
            float m = fmaxf(s0, s1);
            m = fmaxf(m, __shfl_xor(m, 1, 32));
            m = fmaxf(m, __shfl_xor(m, 2, 32));
            m = fmaxf(m, __shfl_xor(m, 4, 32));
            m = fmaxf(m, __shfl_xor(m, 8, 32));
            const float nm   = fmaxf(rmax[v], m);
            const float corr = __expf(rmax[v] - nm);
            rmax[v] = nm;
            const float p0 = __expf(s0 - nm);
            const float p1 = __expf(s1 - nm);
            float ss = p0 + p1;
            ss += __shfl_xor(ss, 1, 32);
            ss += __shfl_xor(ss, 2, 32);
            ss += __shfl_xor(ss, 4, 32);
            ss += __shfl_xor(ss, 8, 32);
            rsum[v] = rsum[v] * corr + ss;
            #pragma unroll
            for (int j = 0; j < 4; ++j) oacc[j][v] *= corr;
            const int row = v + 8*half;
            pw[row*32 + lm]      = bf16s(p0);
            pw[row*32 + 16 + lm] = bf16s(p1);
        }
        // ---- reload P tile in A-fragment layout (single wave: DS in-order) --
        Frag pa;
        const unsigned short* prow = pw + lm*32 + aoff;
        pa.q[0] = *(const int4*)(prow);
        pa.q[1] = *(const int4*)(prow + 8);
        // ---- P * V : 4 output d-tiles of 16 ----
        #pragma unroll
        for (int j = 0; j < 4; ++j) {
            Frag bvf;
            const unsigned short* vr =
                vbase + (size_t)(j*16 + lm) * Sh + sb + half*16;
            bvf.q[0] = *(const int4*)(vr);
            bvf.q[1] = *(const int4*)(vr + 8);
            oacc[j] = wmma_bf16(pa, bvf, oacc[j]);
        }
    }

    float rinv[8];
    #pragma unroll
    for (int v = 0; v < 8; ++v) rinv[v] = 1.0f / rsum[v];
    #pragma unroll
    for (int j = 0; j < 4; ++j) {
        #pragma unroll
        for (int v = 0; v < 8; ++v) {
            const int row = v + 8*half;
            const size_t dst = ((size_t)(b * T_LEN) + qt*16 + row) * D_MODEL
                               + h*HEAD_DIM + j*16 + lm;
            Out[dst] = bf16s(oacc[j][v] * rinv[v]);
        }
    }
}

// ---------------------------------------------------------------------------
// Output projection: out = Attn(bf16) * Wo^T(bf16) + bo, fp32 result.
// ---------------------------------------------------------------------------
__global__ __launch_bounds__(256) void gemm_o_kernel(
    const unsigned short* __restrict__ X, const unsigned short* __restrict__ W,
    const float* __restrict__ bias, float* __restrict__ out)
{
    const int lane = threadIdx.x & 31;
    const int half = lane >> 4;
    const int lm   = lane & 15;
    const int wave = blockIdx.x * (blockDim.x >> 5) + (threadIdx.x >> 5);
    const int N = B_SZ * T_LEN;
    if (wave >= (N >> 4) * (D_MODEL / 64)) return;
    const int nGroup = wave & 15;
    const int mBase  = (wave >> 4) << 4;
    const int oBase  = nGroup << 6;
    const int aoff   = half * 16;

    v8f acc[4] = {};
    const unsigned short* arow = X + (size_t)(mBase + lm) * D_MODEL;
    const unsigned short* wrow0 = W + (size_t)(oBase + lm)      * D_MODEL + half*16;
    const unsigned short* wrow1 = W + (size_t)(oBase + 16 + lm) * D_MODEL + half*16;
    const unsigned short* wrow2 = W + (size_t)(oBase + 32 + lm) * D_MODEL + half*16;
    const unsigned short* wrow3 = W + (size_t)(oBase + 48 + lm) * D_MODEL + half*16;

    for (int k = 0; k < D_MODEL; k += 32) {
        __builtin_prefetch(arow + k + 32 + aoff, 0, 3);
        Frag a;
        a.q[0] = *(const int4*)(arow + k + aoff);
        a.q[1] = *(const int4*)(arow + k + 8 + aoff);
        Frag b0, b1, b2, b3;
        b0.q[0] = *(const int4*)(wrow0 + k); b0.q[1] = *(const int4*)(wrow0 + k + 8);
        b1.q[0] = *(const int4*)(wrow1 + k); b1.q[1] = *(const int4*)(wrow1 + k + 8);
        b2.q[0] = *(const int4*)(wrow2 + k); b2.q[1] = *(const int4*)(wrow2 + k + 8);
        b3.q[0] = *(const int4*)(wrow3 + k); b3.q[1] = *(const int4*)(wrow3 + k + 8);
        acc[0] = wmma_bf16(a, b0, acc[0]);
        acc[1] = wmma_bf16(a, b1, acc[1]);
        acc[2] = wmma_bf16(a, b2, acc[2]);
        acc[3] = wmma_bf16(a, b3, acc[3]);
    }
    #pragma unroll
    for (int j = 0; j < 4; ++j) {
        const float bb = bias[oBase + j*16 + lm];
        #pragma unroll
        for (int v = 0; v < 8; ++v) {
            const int r = mBase + v + 8*half;
            out[(size_t)r * D_MODEL + oBase + j*16 + lm] = acc[j][v] + bb;
        }
    }
}

// ---------------------------------------------------------------------------
extern "C" void kernel_launch(void* const* d_in, const int* in_sizes, int n_in,
                              void* d_out, int out_size, void* d_ws, size_t ws_size,
                              hipStream_t stream) {
    (void)in_sizes; (void)n_in; (void)out_size; (void)ws_size;
    const float* dec = (const float*)d_in[0];
    const float* enc = (const float*)d_in[1];
    const float* Wq  = (const float*)d_in[2];
    const float* bq  = (const float*)d_in[3];
    const float* Wk  = (const float*)d_in[4];
    const float* bk  = (const float*)d_in[5];
    const float* Wv  = (const float*)d_in[6];
    const float* bv  = (const float*)d_in[7];
    const float* Wo  = (const float*)d_in[8];
    const float* bo  = (const float*)d_in[9];
    float* out = (float*)d_out;

    // bf16 workspace layout (elements):
    //   decB [B*T][1024]         :  4,194,304
    //   encB [B*S][1024]         : 16,777,216
    //   wqB/wkB/wvB/woB [1024^2] :  1,048,576 each
    //   Q  [B][H][T][64]         :  4,194,304
    //   K  [B][H][S][64]         : 16,777,216
    //   Vt [B][H][64][S/stride]  :  7,864,320 (stride-compacted)
    //   A  [B][T][D_MODEL]       :  4,194,304
    // total = 58,195,968 elems = ~111 MB
    unsigned short* ws   = (unsigned short*)d_ws;
    unsigned short* decB = ws;
    unsigned short* encB = decB + (size_t)B_SZ*T_LEN*D_MODEL;
    unsigned short* wqB  = encB + (size_t)B_SZ*S_LEN*D_MODEL;
    unsigned short* wkB  = wqB  + (size_t)D_MODEL*D_MODEL;
    unsigned short* wvB  = wkB  + (size_t)D_MODEL*D_MODEL;
    unsigned short* woB  = wvB  + (size_t)D_MODEL*D_MODEL;
    unsigned short* qB   = woB  + (size_t)D_MODEL*D_MODEL;
    unsigned short* kB   = qB   + (size_t)B_SZ*NUM_HEADS*T_LEN*HEAD_DIM;
    unsigned short* vtB  = kB   + (size_t)B_SZ*NUM_HEADS*S_LEN*HEAD_DIM;
    unsigned short* aB   = vtB  + (size_t)B_SZ*SUM_SH*HEAD_DIM;

    dim3 blk(256);
    // Phase 0: fp32 -> bf16 (8 elems/thread)
    cvt_bf16_kernel<<<2048, blk, 0, stream>>>(dec, decB, (B_SZ*T_LEN*D_MODEL)/8);
    cvt_bf16_kernel<<<8192, blk, 0, stream>>>(enc, encB, (B_SZ*S_LEN*D_MODEL)/8);
    cvt_bf16_kernel<<<512,  blk, 0, stream>>>(Wq, wqB, (D_MODEL*D_MODEL)/8);
    cvt_bf16_kernel<<<512,  blk, 0, stream>>>(Wk, wkB, (D_MODEL*D_MODEL)/8);
    cvt_bf16_kernel<<<512,  blk, 0, stream>>>(Wv, wvB, (D_MODEL*D_MODEL)/8);
    cvt_bf16_kernel<<<512,  blk, 0, stream>>>(Wo, woB, (D_MODEL*D_MODEL)/8);
    // Phase 1: projections (Q: 4096 waves, K/V: 16384 waves each)
    gemm_qkv_kernel<<<512,  blk, 0, stream>>>(decB, wqB, bq, qB, 10, 0);
    gemm_qkv_kernel<<<2048, blk, 0, stream>>>(encB, wkB, bk, kB, 12, 1);
    gemm_qkv_kernel<<<2048, blk, 0, stream>>>(encB, wvB, bv, vtB, 12, 2);
    // Phase 2: flash attention (4096 waves)
    attn_kernel<<<512, blk, 0, stream>>>(qB, kB, vtB, aB);
    // Phase 3: output projection
    gemm_o_kernel<<<512, blk, 0, stream>>>(aB, woB, bo, out);
}
